// ActorNetSpiking_52587579572808
// MI455X (gfx1250) — compile-verified, matrix operand order checked by reading, not run on hardware
//
#include <hip/hip_runtime.h>

#define B_ 4096
#define S_ 512
#define T_ 50
#define H_ 256
#define A_ 2
#define BT 32
#define BLK 512           // 16 waves; wave w owns cols [16w,16w+16) x rowtiles {0,1}
#define RT 2              // row-tiles per wave (B-fragment reuse factor)
#define SPK_LD 264        // 256 + 8 f16 pad -> bank rotation per row
#define XT_LD 520         // 512 + 8 f16 pad (matches TDM pad below)

typedef _Float16 v8h  __attribute__((ext_vector_type(8)));
typedef _Float16 v16h __attribute__((ext_vector_type(16)));
typedef _Float16 h2   __attribute__((ext_vector_type(2)));
typedef float    v8f  __attribute__((ext_vector_type(8)));

#if defined(__gfx1250__) && __has_builtin(__builtin_amdgcn_tensor_load_to_lds)
#define HAVE_TDM 1
typedef unsigned int u32x4 __attribute__((ext_vector_type(4)));
typedef int i32x4 __attribute__((ext_vector_type(4)));
typedef int i32x8 __attribute__((ext_vector_type(8)));
#else
#define HAVE_TDM 0
#endif

static __device__ __forceinline__ v16h cat8(v8h lo, v8h hi) {
  return __builtin_shufflevector(lo, hi, 0,1,2,3,4,5,6,7,8,9,10,11,12,13,14,15);
}

static __device__ __forceinline__ v16h ldfrag(const _Float16* p) {
  return cat8(*(const v8h*)p, *(const v8h*)(p + 16));
}

static __device__ __forceinline__ v8f wmma16(v16h a, v16h b, v8f c) {
  return __builtin_amdgcn_wmma_f32_16x16x32_f16(false, a, false, b, (short)0, c,
                                                false, false);
}

#if HAVE_TDM
// DMA one contiguous [32 x 512] f16 tile (32KB) global->LDS, padding 4 dwords
// after every 256 dwords (one row) so the destination matches [32][XT_LD].
static __device__ __forceinline__ void tdm_load_tile(const _Float16* gsrc,
                                                     void* ldst) {
  const unsigned long long ga = (unsigned long long)(uintptr_t)gsrc;
  const unsigned int la = (unsigned int)(uintptr_t)ldst;  // LDS byte offset
  u32x4 g0;
  g0[0] = 1u;                                     // count=1, user mode
  g0[1] = la;                                     // lds_addr
  g0[2] = (unsigned int)(ga & 0xFFFFFFFFu);       // global_addr[31:0]
  g0[3] = (unsigned int)((ga >> 32) & 0x01FFFFFFu) | (2u << 30);  // type=2
  i32x8 g1;
  // data_size=2B | pad_enable | pad_interval=256dw (7) | pad_amount=4dw (3)
  g1[0] = (int)((1u << 16) | (1u << 20) | (7u << 22) | (3u << 25));
  g1[1] = (int)(16384u << 16);   // tensor_dim0 lo16 in [31:16]
  g1[2] = (int)(1u << 16);       // tensor_dim0 hi=0 ; tensor_dim1 lo=1
  g1[3] = (int)(16384u << 16);   // tensor_dim1 hi=0 ; tile_dim0=16384
  g1[4] = 1;                     // tile_dim1=1, tile_dim2=0
  g1[5] = 16384;                 // tensor_dim0_stride lo
  g1[6] = 0;
  g1[7] = 0;
  i32x4 z4 = {0, 0, 0, 0};
#if __clang_major__ >= 23
  i32x8 z8 = {0, 0, 0, 0, 0, 0, 0, 0};
  __builtin_amdgcn_tensor_load_to_lds(g0, g1, z4, z4, z8, 0);
#else
  __builtin_amdgcn_tensor_load_to_lds(g0, g1, z4, z4, 0);
#endif
}
#endif

// Streaming-B GEMM: B fragment loaded once per k-step, reused for both
// row-tiles.  Wn already points at row `ncol` of W.
template <int NKS>
static __device__ __forceinline__ void gemm_stream(const _Float16* __restrict__ Ar0,
                                                   const _Float16* __restrict__ Ar1,
                                                   const _Float16* __restrict__ Wn,
                                                   int half8, v8f (&acc)[RT]) {
  #pragma unroll
  for (int ks = 0; ks < NKS; ++ks) {
    const int k0 = ks * 32 + half8;
    v16h b  = ldfrag(Wn + k0);
    v16h a0 = ldfrag(Ar0 + k0);
    acc[0] = wmma16(a0, b, acc[0]);
    v16h a1 = ldfrag(Ar1 + k0);
    acc[1] = wmma16(a1, b, acc[1]);
  }
}

// Register-B GEMM: B fragments pinned in VGPRs (zero weight loads).
static __device__ __forceinline__ void gemm_breg(const _Float16* __restrict__ Ar0,
                                                 const _Float16* __restrict__ Ar1,
                                                 const v16h (&wb)[8], int half8,
                                                 v8f (&acc)[RT]) {
  #pragma unroll
  for (int ks = 0; ks < 8; ++ks) {
    const int k0 = ks * 32 + half8;
    v16h a0 = ldfrag(Ar0 + k0);
    acc[0] = wmma16(a0, wb[ks], acc[0]);
    v16h a1 = ldfrag(Ar1 + k0);
    acc[1] = wmma16(a1, wb[ks], acc[1]);
  }
}

// Fused LIF update + spike store (bias folded in here; accs start at zero so
// the first WMMA of each chain uses the inline-0 SRC2 form).
static __device__ __forceinline__ void lif_store(const v8f (&acc)[RT],
                                                 float bias, h2 (&st)[RT][8],
                                                 _Float16 (*buf)[SPK_LD],
                                                 int half, int ncol) {
  #pragma unroll
  for (int rt = 0; rt < RT; ++rt) {
    #pragma unroll
    for (int r = 0; r < 8; ++r) {
      h2 s = st[rt][r];
      float syn = (float)s.x;
      float mem = (float)s.y;
      float rst = (mem > 1.0f) ? 1.0f : 0.0f;   // reset from PREVIOUS mem
      syn = 0.9f  * syn + (acc[rt][r] + bias);
      mem = 0.85f * mem + syn - rst;
      s.x = (_Float16)syn;
      s.y = (_Float16)mem;
      st[rt][r] = s;
      buf[rt * 16 + r + 8 * half][ncol] =
          (mem > 1.0f) ? (_Float16)1.0f : (_Float16)0.0f;
    }
  }
}

__global__ void __launch_bounds__(BLK)
__attribute__((amdgpu_waves_per_eu(4, 4)))  // 16-wave WG per WGP: allow the
                                            // full register budget, no spills
snn_main(
    const _Float16* __restrict__ xT,    // [T,B,S] f16
    const _Float16* __restrict__ W1h,   // [256,512] f16 row-major
    const _Float16* __restrict__ W2h,   // [256,256]
    const _Float16* __restrict__ W3h,   // [256,256]
    const float* __restrict__ b1, const float* __restrict__ b2,
    const float* __restrict__ b3,
    const float* __restrict__ W4, const float* __restrict__ b4,
    float* __restrict__ out) {
  __shared__ _Float16 sSpk[2][BT][SPK_LD];   // ping-pong spike buffers
#if HAVE_TDM
  __shared__ _Float16 sXt[2][BT][XT_LD];
#endif
  __shared__ float sW4[A_][H_];
  __shared__ float sb4[A_];

  const int tid   = threadIdx.x;
  const int lane  = tid & 31;
  const int wid   = tid >> 5;            // 16 waves, one 16-col tile each
  const int half  = (lane >> 4) & 1;
  const int half8 = half * 8;
  const int lrow  = lane & 15;
  const int ncol  = wid * 16 + lrow;     // this lane's output column
  const int wgB0  = blockIdx.x * BT;

  for (int i = tid; i < A_ * H_; i += BLK) sW4[i >> 8][i & (H_ - 1)] = W4[i];
  if (tid < A_) sb4[tid] = b4[tid];

  // --- pin W2/W3 slices (16 cols x 256) in VGPRs as WMMA B fragments ---
  v16h wB2[8], wB3[8];
  #pragma unroll
  for (int ks = 0; ks < 8; ++ks) {
    const int k0 = ks * 32 + half8;
    wB2[ks] = ldfrag(W2h + (size_t)ncol * H_ + k0);
    wB3[ks] = ldfrag(W3h + (size_t)ncol * H_ + k0);
  }

  // recurrent LIF state (syn, mem) packed 2xf16, resident in VGPRs
  h2 st1[RT][8], st2[RT][8], st3[RT][8];
  h2 z; z.x = (_Float16)0.0f; z.y = (_Float16)0.0f;
  #pragma unroll
  for (int rt = 0; rt < RT; ++rt)
    #pragma unroll
    for (int r = 0; r < 8; ++r) { st1[rt][r] = z; st2[rt][r] = z; st3[rt][r] = z; }

  const float bias1 = b1[ncol], bias2 = b2[ncol], bias3 = b3[ncol];

  float syn4 = 0.0f, mem4 = 0.0f, accO = 0.0f;
  const int m4 = tid >> 1, a4 = tid & 1;

#if HAVE_TDM
  if (wid == 0) tdm_load_tile(xT + (size_t)wgB0 * S_, &sXt[0][0][0]);
#endif

  for (int t = 0; t < T_; ++t) {
#if HAVE_TDM
    if (wid == 0) __builtin_amdgcn_s_wait_tensorcnt(0);  // tile t landed
#endif
    __syncthreads();  // tile ready; layer-4 readers of sSpk[0] (t-1) done
#if HAVE_TDM
    if (wid == 0 && t + 1 < T_)
      tdm_load_tile(xT + (size_t)(t + 1) * ((size_t)B_ * S_) +
                        (size_t)wgB0 * S_,
                    &sXt[(t + 1) & 1][0][0]);
#endif

    // Launder a *uniform integer offset* (not the pointer) so LICM cannot
    // hoist the t-invariant W1 loads, while the address stays derived from
    // the __restrict__ global kernel arg -> global_load_b128 (not flat).
    size_t woff = 0;
    asm volatile("" : "+s"(woff));
    const _Float16* W1t = W1h + (size_t)ncol * S_ + woff;

    // ---------------- layer 1: [32,512] x [512,256] -> sSpk[0] -----------
    v8f acc[RT] = {};   // zero C: first WMMA uses inline-0 SRC2

#if HAVE_TDM
    gemm_stream<16>(&sXt[t & 1][lrow][0], &sXt[t & 1][16 + lrow][0], W1t,
                    half8, acc);
#else
    const _Float16* aBase =
        xT + (size_t)t * ((size_t)B_ * S_) + (size_t)(wgB0 + lrow) * S_;
    gemm_stream<16>(aBase, aBase + 16 * S_, W1t, half8, acc);
    if (t + 1 < T_) __builtin_prefetch(aBase + (size_t)B_ * S_, 0, 1);
#endif
    lif_store(acc, bias1, st1, sSpk[0], half, ncol);
    __syncthreads();

    // ---------------- layer 2: sSpk[0] x W2 -> sSpk[1], B in VGPRs -------
    v8f acc2[RT] = {};
    gemm_breg(&sSpk[0][lrow][0], &sSpk[0][16 + lrow][0], wB2, half8, acc2);
    lif_store(acc2, bias2, st2, sSpk[1], half, ncol);
    __syncthreads();

    // ---------------- layer 3: sSpk[1] x W3 -> sSpk[0], B in VGPRs -------
    v8f acc3[RT] = {};
    gemm_breg(&sSpk[1][lrow][0], &sSpk[1][16 + lrow][0], wB3, half8, acc3);
    lif_store(acc3, bias3, st3, sSpk[0], half, ncol);
    __syncthreads();

    // ---------------- layer 4: [32,256] x [256,2] (VALU) ----------------
    if (tid < BT * A_) {
      float cur = sb4[a4];
      const h2* srow = (const h2*)&sSpk[0][m4][0];
      const float* w = &sW4[a4][0];
      #pragma unroll 8
      for (int k = 0; k < H_ / 2; ++k) {
        h2 sp = srow[k];
        cur += (float)sp.x * w[2 * k] + (float)sp.y * w[2 * k + 1];
      }
      const float rst = (mem4 > 1.0f) ? 1.0f : 0.0f;
      syn4 = 0.9f * syn4 + cur;
      mem4 = 0.85f * mem4 + syn4 - rst;
      accO += (mem4 > 1.0f) ? 1.0f : 0.0f;
    }
    // loop-top barrier guards sSpk[0] reuse
  }

  if (tid < BT * A_)
    out[(size_t)(wgB0 + m4) * A_ + a4] = accO * (1.0f / (float)T_);
}

// ---- plain f32 fallback (only when workspace is too small): 1 block/row ----
__global__ void __launch_bounds__(256) snn_fallback(
    const float* __restrict__ x, const float* __restrict__ W1,
    const float* __restrict__ b1, const float* __restrict__ W2,
    const float* __restrict__ b2, const float* __restrict__ W3,
    const float* __restrict__ b3, const float* __restrict__ W4,
    const float* __restrict__ b4, float* __restrict__ out) {
  __shared__ float sxt[S_];
  __shared__ float sspk[H_];
  const int b = blockIdx.x;
  const int n = threadIdx.x;
  float syn1 = 0, mem1 = 0, syn2 = 0, mem2 = 0, syn3 = 0, mem3 = 0;
  float syn4 = 0, mem4 = 0, accO = 0;
  for (int t = 0; t < T_; ++t) {
    __syncthreads();
    for (int s = n; s < S_; s += 256) sxt[s] = x[((size_t)b * S_ + s) * T_ + t];
    __syncthreads();
    float cur = b1[n];
    #pragma unroll 4
    for (int k = 0; k < S_; ++k) cur += sxt[k] * W1[n * S_ + k];
    float rst = (mem1 > 1.0f) ? 1.0f : 0.0f;
    syn1 = 0.9f * syn1 + cur; mem1 = 0.85f * mem1 + syn1 - rst;
    sspk[n] = (mem1 > 1.0f) ? 1.0f : 0.0f;
    __syncthreads();
    cur = b2[n];
    #pragma unroll 4
    for (int k = 0; k < H_; ++k) cur += sspk[k] * W2[n * H_ + k];
    __syncthreads();
    rst = (mem2 > 1.0f) ? 1.0f : 0.0f;
    syn2 = 0.9f * syn2 + cur; mem2 = 0.85f * mem2 + syn2 - rst;
    sspk[n] = (mem2 > 1.0f) ? 1.0f : 0.0f;
    __syncthreads();
    cur = b3[n];
    #pragma unroll 4
    for (int k = 0; k < H_; ++k) cur += sspk[k] * W3[n * H_ + k];
    __syncthreads();
    rst = (mem3 > 1.0f) ? 1.0f : 0.0f;
    syn3 = 0.9f * syn3 + cur; mem3 = 0.85f * mem3 + syn3 - rst;
    sspk[n] = (mem3 > 1.0f) ? 1.0f : 0.0f;
    __syncthreads();
    if (n < A_) {
      cur = b4[n];
      #pragma unroll 4
      for (int k = 0; k < H_; ++k) cur += sspk[k] * W4[n * H_ + k];
      rst = (mem4 > 1.0f) ? 1.0f : 0.0f;
      syn4 = 0.9f * syn4 + cur; mem4 = 0.85f * mem4 + syn4 - rst;
      accO += (mem4 > 1.0f) ? 1.0f : 0.0f;
    }
  }
  if (n < A_) out[(size_t)b * A_ + n] = accO * (1.0f / (float)T_);
}

// ---- prep: weights f32 -> f16 (row-major preserved) ----
__global__ void cvt_weights(const float* __restrict__ W1,
                            const float* __restrict__ W2,
                            const float* __restrict__ W3,
                            _Float16* __restrict__ W1h,
                            _Float16* __restrict__ W2h,
                            _Float16* __restrict__ W3h) {
  const int i = blockIdx.x * blockDim.x + threadIdx.x;
  if (i < H_ * S_) W1h[i] = (_Float16)W1[i];
  const int j = i - H_ * S_;
  if (j >= 0 && j < H_ * H_) W2h[j] = (_Float16)W2[j];
  const int k = i - (H_ * S_ + H_ * H_);
  if (k >= 0 && k < H_ * H_) W3h[k] = (_Float16)W3[k];
}

// ---- prep: x [B,S,T] f32 -> xT [T,B,S] f16 ----
__global__ void transpose_x(const float* __restrict__ x,
                            _Float16* __restrict__ xT) {
  const size_t i = (size_t)blockIdx.x * blockDim.x + threadIdx.x;  // b*S + s
  if (i >= (size_t)B_ * S_) return;
  const float* p = x + i * T_;
  #pragma unroll
  for (int t = 0; t < T_; ++t)
    xT[(size_t)t * ((size_t)B_ * S_) + i] = (_Float16)p[t];
}

extern "C" void kernel_launch(void* const* d_in, const int* in_sizes, int n_in,
                              void* d_out, int out_size, void* d_ws,
                              size_t ws_size, hipStream_t stream) {
  const float* x  = (const float*)d_in[0];
  const float* W1 = (const float*)d_in[1];
  const float* b1 = (const float*)d_in[2];
  const float* W2 = (const float*)d_in[3];
  const float* b2 = (const float*)d_in[4];
  const float* W3 = (const float*)d_in[5];
  const float* b3 = (const float*)d_in[6];
  const float* W4 = (const float*)d_in[7];
  const float* b4 = (const float*)d_in[8];
  float* out = (float*)d_out;

  char* ws = (char*)d_ws;
  const size_t szW1 = (size_t)H_ * S_ * sizeof(_Float16);       // 256 KB
  const size_t szW2 = (size_t)H_ * H_ * sizeof(_Float16);       // 128 KB
  const size_t wOff = szW1 + 2 * szW2;                          // 512 KB
  const size_t szXT = (size_t)T_ * B_ * S_ * sizeof(_Float16);  // ~200 MB

  if (ws_size < wOff + szXT) {
    // workspace too small for the staged path: plain f32 kernel
    snn_fallback<<<B_, 256, 0, stream>>>(x, W1, b1, W2, b2, W3, b3, W4, b4,
                                         out);
    return;
  }

  _Float16* W1h = (_Float16*)ws;
  _Float16* W2h = (_Float16*)(ws + szW1);
  _Float16* W3h = (_Float16*)(ws + szW1 + szW2);
  _Float16* xT  = (_Float16*)(ws + wOff);

  cvt_weights<<<(H_ * S_ + 2 * H_ * H_ + 255) / 256, 256, 0, stream>>>(
      W1, W2, W3, W1h, W2h, W3h);
  transpose_x<<<(B_ * S_ + 255) / 256, 256, 0, stream>>>(x, xT);
  snn_main<<<B_ / BT, BLK, 0, stream>>>(xT, W1h, W2h, W3h, b1, b2, b3, W4, b4,
                                        out);
}
